// VectorQuantizer_11759620456816
// MI455X (gfx1250) — matrix operand order repaired and checked by reading, not compiled, not run on hardware
//
#include <hip/hip_runtime.h>
#include <hip/hip_bf16.h>

typedef __attribute__((ext_vector_type(16))) _Float16 v16h;
typedef __attribute__((ext_vector_type(8)))  float    v8f;

#define N_ROWS   65536          // 64*32*32
#define DIM      256
#define K_CODES  1024
#define N_TILES  (N_ROWS / 16)  // 4096 row-tiles of 16
#define QN       (N_ROWS * DIM) // quantized element count

// ---------------------------------------------------------------------------
// Kernel 1: pack codebook into WMMA B-matrix register layout (f16 hi/lo split)
// and compute ||c_k||^2.
// B layout used here with v_wmma_f32_16x16x32_f16 (32x16, K x N, contiguous K), per 32x16 tile:
//   lane n   (0..15) : halves 0..15 = K 0..15  of column n
//   lane n+16        : halves 0..15 = K 16..31 of column n
// Stored as [ntile(64)][ktile(8)][lane(32)][half(16)] f16 (512 halves/tile,
// 4096 halves = 8KB per ntile per plane).
// ---------------------------------------------------------------------------
__global__ void __launch_bounds__(256) vq_pack_codebook(
    const float* __restrict__ cb, _Float16* __restrict__ bhi,
    _Float16* __restrict__ blo, float* __restrict__ cnorm) {
  int n = blockIdx.x;       // code index 0..1023
  int k = threadIdx.x;      // dim index  0..255
  float x = cb[(size_t)n * DIM + k];
  _Float16 h = (_Float16)x;
  _Float16 l = (_Float16)(x - (float)h);
  int ntile = n >> 4, nlane = n & 15;
  int ktile = k >> 5, kk = k & 31;
  int lane  = nlane + ((kk >= 16) ? 16 : 0);
  size_t dst = ((size_t)(ntile * 8 + ktile)) * 512 + (size_t)lane * 16 + (kk & 15);
  bhi[dst] = h;
  blo[dst] = l;

  __shared__ float s[256];
  s[k] = x * x;
  __syncthreads();
  for (int st = 128; st > 0; st >>= 1) {
    if (k < st) s[k] += s[k + st];
    __syncthreads();
  }
  if (k == 0) cnorm[n] = s[0];
}

// ---------------------------------------------------------------------------
// Kernel 2: main VQ kernel. One wave32 per 16-row tile, 8 waves per block.
// B tiles staged once per block into LDS via async global->LDS DMA
// (double-buffered, ASYNCcnt-tracked), consumed by all 8 waves via ds_load.
//   dist = ||c||^2 - 2 * (z . c), z.c via 3x f16 WMMA (hi/lo split, f32 acc).
// ---------------------------------------------------------------------------
__global__ void __launch_bounds__(256) vq_main(
    const float* __restrict__ z, const float* __restrict__ cb,
    const _Float16* __restrict__ bhi, const _Float16* __restrict__ blo,
    const float* __restrict__ cnorm, float* __restrict__ outq,
    float* __restrict__ outidx, float* __restrict__ partial) {
  __shared__ __align__(32) _Float16 ldsb[2][2][4096]; // [buf][hi/lo][8KB plane]

  const int wave = threadIdx.x >> 5;
  const int lane = threadIdx.x & 31;
  const int tile = blockIdx.x * 8 + wave;   // 0..4095
  const int r0   = tile * 16;
  const int m    = lane & 15;
  const int kb   = (lane < 16) ? 0 : 16;    // A-layout K base for this lane

  // ---- Async B-stage setup: waves 0-3 copy hi plane, waves 4-7 lo plane ----
  const int   plane  = wave >> 2;           // 0 = hi, 1 = lo
  const int   w2     = wave & 3;            // 2KB chunk within plane
  const _Float16* gplane = plane ? blo : bhi;
  // each async b128 moves 16B/lane * 32 lanes = 512B; 4 per wave = 2KB chunk
  auto issue_batch = [&](int nt, int buf) {
    unsigned long long ga = (unsigned long long)(
        gplane + (size_t)nt * 4096 + (size_t)w2 * 1024 + (size_t)lane * 8);
    unsigned lbase = (unsigned)(size_t)(
        (__attribute__((address_space(3))) void*)&ldsb[buf][plane][w2 * 1024 + lane * 8]);
#pragma unroll
    for (int i = 0; i < 4; ++i) {
      unsigned           l = lbase + i * 512;
      unsigned long long g = ga + i * 512;
      asm volatile("global_load_async_to_lds_b128 %0, %1, off"
                   :: "v"(l), "v"(g) : "memory");
    }
  };

  // ---- Build A tiles (16x32 f16, hi/lo) for all 8 k-tiles, contiguous-K A layout ----
  const float* zrow = z + (size_t)(r0 + m) * DIM;
  v16h ahi[8], alo[8];
#pragma unroll
  for (int kt = 0; kt < 8; ++kt) {
    const float4* p0 = (const float4*)(zrow + kt * 32 + kb);       // K kb..kb+7
    const float4* p1 = (const float4*)(zrow + kt * 32 + kb + 8);   // K kb+8..kb+15
    float4 q0 = p0[0], q1 = p0[1], q2 = p1[0], q3 = p1[1];
    float x[16] = {q0.x, q0.y, q0.z, q0.w, q1.x, q1.y, q1.z, q1.w,
                   q2.x, q2.y, q2.z, q2.w, q3.x, q3.y, q3.z, q3.w};
#pragma unroll
    for (int e = 0; e < 16; ++e) {
      _Float16 h = (_Float16)x[e];
      ahi[kt][e] = h;
      alo[kt][e] = (_Float16)(x[e] - (float)h);
    }
  }

  // ---- Sweep all 64 code-tiles with double-buffered async B staging ----
  float bestv[8];
  int   besti[8];
#pragma unroll
  for (int j = 0; j < 8; ++j) { bestv[j] = 3.4e38f; besti[j] = 0; }

  issue_batch(0, 0);   // prologue: batch 0 -> buffer 0

  for (int nt = 0; nt < 64; ++nt) {
    const int buf = nt & 1;
    if (nt + 1 < 64) {
      issue_batch(nt + 1, buf ^ 1);
      asm volatile("s_wait_asynccnt 4" ::: "memory");  // batch nt landed
    } else {
      asm volatile("s_wait_asynccnt 0" ::: "memory");
    }
    __syncthreads();   // all waves' slices of batch nt visible in LDS

    const _Float16* bh = &ldsb[buf][0][lane * 16];
    const _Float16* bl = &ldsb[buf][1][lane * 16];
    v8f acc = {};
#pragma unroll
    for (int kt = 0; kt < 8; ++kt) {
      v16h vbh = *(const v16h*)(bh + kt * 512);
      v16h vbl = *(const v16h*)(bl + kt * 512);
      acc = __builtin_amdgcn_wmma_f32_16x16x32_f16(false, ahi[kt], false, vbh,
                                                   (short)0, acc, false, false);
      acc = __builtin_amdgcn_wmma_f32_16x16x32_f16(false, ahi[kt], false, vbl,
                                                   (short)0, acc, false, false);
      acc = __builtin_amdgcn_wmma_f32_16x16x32_f16(false, alo[kt], false, vbh,
                                                   (short)0, acc, false, false);
    }
    float cn = cnorm[nt * 16 + m];
    int   ci = nt * 16 + m;
#pragma unroll
    for (int j = 0; j < 8; ++j) {
      float d = fmaf(-2.0f, acc[j], cn);
      if (d < bestv[j]) { bestv[j] = d; besti[j] = ci; }
    }
    __syncthreads();   // done reading buf before it is overwritten
  }

  // ---- Argmin across the 16 lanes of each half-wave (first-index ties) ----
#pragma unroll
  for (int j = 0; j < 8; ++j) {
#pragma unroll
    for (int mask = 1; mask < 16; mask <<= 1) {
      float ov = __shfl_xor(bestv[j], mask, 32);
      int   oi = __shfl_xor(besti[j], mask, 32);
      if (ov < bestv[j] || (ov == bestv[j] && oi < besti[j])) {
        bestv[j] = ov; besti[j] = oi;
      }
    }
  }

  const int half = lane >> 4;   // 0: rows 0..7, 1: rows 8..15
  const int sub  = lane & 15;

  if (sub == 0) {
#pragma unroll
    for (int j = 0; j < 8; ++j)
      outidx[r0 + half * 8 + j] = (float)besti[j];
  }

  // ---- Gather codebook rows -> quantized output; accumulate loss ----
  float sumsq = 0.f;
#pragma unroll
  for (int j = 0; j < 8; ++j) {
    size_t r = (size_t)(r0 + half * 8 + j);
    const float* crow = cb + (size_t)besti[j] * DIM;   // uniform per half-wave
    const float* zr   = z + r * DIM;
    float*       qo   = outq + r * DIM;
#pragma unroll
    for (int i = 0; i < 16; i += 4) {
      float4 c4 = *(const float4*)(crow + sub * 16 + i);
      float4 z4 = *(const float4*)(zr + sub * 16 + i);
      *(float4*)(qo + sub * 16 + i) = c4;
      float dx = c4.x - z4.x, dy = c4.y - z4.y;
      float dz = c4.z - z4.z, dw = c4.w - z4.w;
      sumsq += dx * dx + dy * dy + dz * dz + dw * dw;
    }
  }
#pragma unroll
  for (int mask = 1; mask < 32; mask <<= 1)
    sumsq += __shfl_xor(sumsq, mask, 32);
  if (lane == 0) partial[tile] = sumsq;
}

// ---------------------------------------------------------------------------
// Kernel 3: deterministic fixed-order loss reduction (no float atomics).
// ---------------------------------------------------------------------------
__global__ void __launch_bounds__(256) vq_reduce(
    const float* __restrict__ partial, float* __restrict__ out_sc) {
  __shared__ float s[256];
  int t = threadIdx.x;
  float v = 0.f;
  for (int i = t; i < N_TILES; i += 256) v += partial[i];
  s[t] = v;
  __syncthreads();
  for (int st = 128; st > 0; st >>= 1) {
    if (t < st) s[t] += s[t + st];
    __syncthreads();
  }
  if (t == 0) {
    float mean = s[0] / (float)QN;
    out_sc[0] = 0.25f * mean;   // commitment_loss
    out_sc[1] = mean;           // embedding_loss
  }
}

// ---------------------------------------------------------------------------
extern "C" void kernel_launch(void* const* d_in, const int* in_sizes, int n_in,
                              void* d_out, int out_size, void* d_ws, size_t ws_size,
                              hipStream_t stream) {
  const float* z  = (const float*)d_in[0];   // [65536, 256] f32
  const float* cb = (const float*)d_in[1];   // [1024, 256]  f32
  float* out = (float*)d_out;
  // d_out layout: quantized[QN] | commitment | embedding | indices[65536] (as f32)

  // Workspace layout (~1.06 MB): B-layout hi plane | lo plane | cnorm | partials
  _Float16* bhi   = (_Float16*)d_ws;                       // 512 KB
  _Float16* blo   = bhi + (size_t)K_CODES * DIM;           // 512 KB
  float*    cnorm = (float*)(blo + (size_t)K_CODES * DIM); // 4 KB
  float*    part  = cnorm + K_CODES;                       // 16 KB

  vq_pack_codebook<<<K_CODES, 256, 0, stream>>>(cb, bhi, blo, cnorm);
  vq_main<<<N_TILES / 8, 256, 0, stream>>>(z, cb, bhi, blo, cnorm,
                                           out, out + QN + 2, part);
  vq_reduce<<<1, 256, 0, stream>>>(part, out + QN);
}
